// GGSNNModel_89232240542055
// MI455X (gfx1250) — matrix operand order repaired and checked by reading, not compiled, not run on hardware
//
#include <hip/hip_runtime.h>
#include <hip/hip_bf16.h>
#include <stdint.h>

#define N_NODES 100000
#define N_EDGES 640000
#define IN_DIM  64
#define HIDDEN  128
#define N_STEPS 5

typedef __bf16 bf16_t;
typedef __attribute__((ext_vector_type(16))) __bf16 v16bf;
typedef __attribute__((ext_vector_type(8)))  __bf16 v8bf;
typedef __attribute__((ext_vector_type(8)))  float  v8f;

// ---------------------------------------------------------------------------
// WMMA helpers (CDNA5 16x16x32 bf16, fp32 accumulate)
// ---------------------------------------------------------------------------
__device__ __forceinline__ v8f wmma_bf16(v16bf a, v16bf b, v8f c) {
  return __builtin_amdgcn_wmma_f32_16x16x32_bf16(false, a, false, b, (short)0, c,
                                                 false, false);
}

// B-matrix fragment (32x16 KxN, bf16), W row-major [out][in]:
//   col N = nbase+l15; elems 0-7 = K k0+hi*8..+7, elems 8-15 = K k0+hi*8+16..+23.
__device__ __forceinline__ v16bf load_b_frag(const bf16_t* __restrict__ W,
                                             int nbase, int k0, int l15, int hi) {
  const bf16_t* p = W + (size_t)(nbase + l15) * HIDDEN + k0 + hi * 8;
  v8bf lo = *(const v8bf*)(p);
  v8bf hv = *(const v8bf*)(p + 16);
  v16bf f;
#pragma unroll
  for (int i = 0; i < 8; ++i) { f[i] = lo[i]; f[i + 8] = hv[i]; }
  return f;
}

// A-matrix fragment (16x32 MxK, bf16): lane group g: row=l15,
// VGPR0-3 = K k0+g*8..+7, VGPR4-7 = K k0+g*8+16..+23 (two 16B runs).
__device__ __forceinline__ v16bf load_a_frag_bf(const bf16_t* A, int row,
                                                int k0, int hi) {
  const bf16_t* p = A + row * HIDDEN + k0 + hi * 8;
  v8bf lo = *(const v8bf*)(p);
  v8bf hv = *(const v8bf*)(p + 16);
  v16bf f;
#pragma unroll
  for (int i = 0; i < 8; ++i) { f[i] = lo[i]; f[i + 8] = hv[i]; }
  return f;
}

// Same, fp32 source converted on the fly.
__device__ __forceinline__ v16bf load_a_frag_f32(const float* A, int row,
                                                 int k0, int hi) {
  const float* p = A + row * HIDDEN + k0 + hi * 8;
  v8f lo = *(const v8f*)(p);
  v8f hv = *(const v8f*)(p + 16);
  v16bf f;
#pragma unroll
  for (int i = 0; i < 8; ++i) { f[i] = (bf16_t)lo[i]; f[i + 8] = (bf16_t)hv[i]; }
  return f;
}

// Async global->LDS copy of one 16B chunk (ASYNCcnt-tracked, no VGPR roundtrip).
__device__ __forceinline__ void async_copy_b128(const void* gptr, void* lptr) {
  const uint32_t la = (uint32_t)(uintptr_t)lptr;  // addr[31:0] = LDS offset
  asm volatile("global_load_async_to_lds_b128 %0, %1, off" ::"v"(la), "v"(gptr)
               : "memory");
}
__device__ __forceinline__ void wait_async_all() {
  asm volatile("s_wait_asynccnt 0x0" ::: "memory");
}

// ---------------------------------------------------------------------------
// m = h @ W.T + b (bf16 out). 80 rows/block (1250 blocks exact); h-tile staged
// in LDS via async copies. All 5 A-frags of a k-step are fetched before the
// WMMA burst so ds waits overlap with earlier WMMAs.
// ---------------------------------------------------------------------------
__global__ __launch_bounds__(256) void gemm_m_kernel(
    const bf16_t* __restrict__ H, const bf16_t* __restrict__ Wb,
    const float* __restrict__ bias, bf16_t* __restrict__ M) {
  __shared__ __align__(16) bf16_t sH[80 * HIDDEN];  // 20 KB
  const int tid = threadIdx.x;
  const int lane = tid & 31, wave = tid >> 5;
  const int l15 = lane & 15, hi = lane >> 4;
  const int m0 = blockIdx.x * 80;
  const int nb = wave * 16;

  {  // 80*128 bf16 = 1280 x b128 chunks; 5 per thread, contiguous.
    const bf16_t* gH = H + (size_t)m0 * HIDDEN;
#pragma unroll
    for (int i = 0; i < 5; ++i) {
      const int idx = (i * 256 + tid) * 8;
      async_copy_b128(gH + idx, &sH[idx]);
    }
    wait_async_all();
  }
  __syncthreads();

  v8f acc[5] = {};
#pragma unroll
  for (int ks = 0; ks < 4; ++ks) {
    const int k0 = ks * 32;
    v16bf bF = load_b_frag(Wb, nb, k0, l15, hi);
    v16bf aF[5];
#pragma unroll
    for (int mt = 0; mt < 5; ++mt) aF[mt] = load_a_frag_bf(sH, mt * 16 + l15, k0, hi);
#pragma unroll
    for (int mt = 0; mt < 5; ++mt) acc[mt] = wmma_bf16(aF[mt], bF, acc[mt]);
  }
  const float bcol = bias[nb + l15];
#pragma unroll
  for (int mt = 0; mt < 5; ++mt) {
#pragma unroll
    for (int r = 0; r < 8; ++r) {
      const int row = m0 + mt * 16 + hi * 8 + r;
      M[(size_t)row * HIDDEN + nb + l15] = (bf16_t)(acc[mt][r] + bcol);
    }
  }
}

// ---------------------------------------------------------------------------
// a[dst] += m[src] : 16 threads/edge, 8 columns each, native f32 L2 atomics.
// ---------------------------------------------------------------------------
__global__ __launch_bounds__(256) void scatter_kernel(
    const bf16_t* __restrict__ M, const int* __restrict__ src,
    const int* __restrict__ dst, float* __restrict__ A) {
  const int tid = blockIdx.x * 256 + threadIdx.x;  // exactly N_EDGES*16 threads
  const int e = tid >> 4;
  const int c = (tid & 15) * 8;
  const int s = src[e], d = dst[e];
  v8bf mv = *(const v8bf*)(M + (size_t)s * HIDDEN + c);
  float* ap = A + (size_t)d * HIDDEN + c;
#pragma unroll
  for (int i = 0; i < 8; ++i) {
    float v = (float)mv[i];
    asm volatile("global_atomic_add_f32 %0, %1, off" ::"v"(ap + i), "v"(v)
                 : "memory");
  }
}

// ---------------------------------------------------------------------------
// Fused GRU cell: h = GRUCell(a, h), in place. 32 rows/block (3125 exact).
// a (fp32) + h (bf16) tiles staged in LDS asynchronously; wave w owns H
// columns [16w,16w+16). All 4 A-frags and all 6 B-frags of a k-step are
// issued before the 12-WMMA burst.
// ---------------------------------------------------------------------------
__global__ __launch_bounds__(256) void gru_kernel(
    const float* __restrict__ A, bf16_t* __restrict__ H,
    const bf16_t* __restrict__ Wih, const bf16_t* __restrict__ Whh,
    const float* __restrict__ bih, const float* __restrict__ bhh) {
  __shared__ __align__(16) float  sA[32 * HIDDEN];  // 16 KB
  __shared__ __align__(16) bf16_t sH[32 * HIDDEN];  //  8 KB
  const int tid = threadIdx.x;
  const int lane = tid & 31, wave = tid >> 5;
  const int l15 = lane & 15, hi = lane >> 4;
  const int m0 = blockIdx.x * 32;
  const int col = wave * 16 + l15;  // this lane's H column

  {  // stage contiguous full-width tiles: a 16 KB, h 8 KB
    const float* gA = A + (size_t)m0 * HIDDEN;
#pragma unroll
    for (int i = 0; i < 4; ++i) {
      const int idx = (i * 256 + tid) * 4;
      async_copy_b128(gA + idx, &sA[idx]);
    }
    const bf16_t* gH = H + (size_t)m0 * HIDDEN;
#pragma unroll
    for (int i = 0; i < 2; ++i) {
      const int idx = (i * 256 + tid) * 8;
      async_copy_b128(gH + idx, &sH[idx]);
    }
    wait_async_all();
  }
  __syncthreads();

  v8f acc[6][2] = {};
#pragma unroll
  for (int ks = 0; ks < 4; ++ks) {
    const int k0 = ks * 32;
    v16bf bI[3], bH[3];
#pragma unroll
    for (int t = 0; t < 3; ++t) {
      bI[t] = load_b_frag(Wih, t * HIDDEN + wave * 16, k0, l15, hi);
      bH[t] = load_b_frag(Whh, t * HIDDEN + wave * 16, k0, l15, hi);
    }
    v16bf aF[2], hF[2];
#pragma unroll
    for (int mt = 0; mt < 2; ++mt) {
      aF[mt] = load_a_frag_f32(sA, mt * 16 + l15, k0, hi);
      hF[mt] = load_a_frag_bf(sH, mt * 16 + l15, k0, hi);
    }
#pragma unroll
    for (int t = 0; t < 3; ++t) {
#pragma unroll
      for (int mt = 0; mt < 2; ++mt) {
        acc[t][mt]     = wmma_bf16(aF[mt], bI[t], acc[t][mt]);
        acc[3 + t][mt] = wmma_bf16(hF[mt], bH[t], acc[3 + t][mt]);
      }
    }
  }

  const float bir = bih[col], biz = bih[HIDDEN + col], bin_ = bih[2 * HIDDEN + col];
  const float bhr = bhh[col], bhz = bhh[HIDDEN + col], bhn  = bhh[2 * HIDDEN + col];
#pragma unroll
  for (int mt = 0; mt < 2; ++mt) {
#pragma unroll
    for (int r = 0; r < 8; ++r) {
      const int lrow = mt * 16 + hi * 8 + r;
      const float ir = acc[0][mt][r] + bir;
      const float iz = acc[1][mt][r] + biz;
      const float in_ = acc[2][mt][r] + bin_;
      const float hr = acc[3][mt][r] + bhr;
      const float hz = acc[4][mt][r] + bhz;
      const float hn = acc[5][mt][r] + bhn;
      const float rg = 1.f / (1.f + __expf(-(ir + hr)));
      const float zg = 1.f / (1.f + __expf(-(iz + hz)));
      const float ng = tanhf(in_ + rg * hn);
      const float hprev = (float)sH[lrow * HIDDEN + col];
      H[(size_t)(m0 + lrow) * HIDDEN + col] = (bf16_t)((1.f - zg) * ng + zg * hprev);
    }
  }
}

// ---------------------------------------------------------------------------
// Small helpers
// ---------------------------------------------------------------------------
__global__ __launch_bounds__(256) void zero_kernel(float4* __restrict__ p) {
  p[blockIdx.x * 256 + threadIdx.x] = float4{0.f, 0.f, 0.f, 0.f};
}

__global__ __launch_bounds__(256) void pad_kernel(const float* __restrict__ F,
                                                  bf16_t* __restrict__ H) {
  const int tid = blockIdx.x * 256 + threadIdx.x;  // N_NODES*128 exact
  const int n = tid >> 7, k = tid & 127;
  H[tid] = (k < IN_DIM) ? (bf16_t)F[n * IN_DIM + k] : (bf16_t)0.f;
}

__global__ __launch_bounds__(256) void cvt_kernel(const float* __restrict__ S,
                                                  bf16_t* __restrict__ D, int n) {
  const int tid = blockIdx.x * 256 + threadIdx.x;
  if (tid < n) D[tid] = (bf16_t)S[tid];
}

__global__ __launch_bounds__(256) void head_kernel(
    const bf16_t* __restrict__ H, const float* __restrict__ fw,
    const float* __restrict__ fb, float* __restrict__ out) {
  const int n = blockIdx.x * 256 + threadIdx.x;
  if (n >= N_NODES) return;
  float a0 = fb[0], a1 = fb[1];
  const bf16_t* hr = H + (size_t)n * HIDDEN;
#pragma unroll 8
  for (int k = 0; k < HIDDEN; ++k) {
    const float hv = (float)hr[k];
    a0 += hv * fw[k];
    a1 += hv * fw[HIDDEN + k];
  }
  out[n * 2 + 0] = a0;
  out[n * 2 + 1] = a1;
}

// ---------------------------------------------------------------------------
// Host launch
// ---------------------------------------------------------------------------
extern "C" void kernel_launch(void* const* d_in, const int* in_sizes, int n_in,
                              void* d_out, int out_size, void* d_ws,
                              size_t ws_size, hipStream_t stream) {
  (void)in_sizes; (void)n_in; (void)out_size; (void)ws_size;
  const float* features = (const float*)d_in[0];
  const int* src = (const int*)d_in[1];
  const int* dst = (const int*)d_in[2];
  const float* W[2]   = {(const float*)d_in[3],  (const float*)d_in[9]};
  const float* b[2]   = {(const float*)d_in[4],  (const float*)d_in[10]};
  const float* wih[2] = {(const float*)d_in[5],  (const float*)d_in[11]};
  const float* whh[2] = {(const float*)d_in[6],  (const float*)d_in[12]};
  const float* bih[2] = {(const float*)d_in[7],  (const float*)d_in[13]};
  const float* bhh[2] = {(const float*)d_in[8],  (const float*)d_in[14]};
  const float* fc_w = (const float*)d_in[15];
  const float* fc_b = (const float*)d_in[16];

  char* ws = (char*)d_ws;
  size_t off = 0;
  auto alloc = [&](size_t bytes) -> void* {
    void* p = ws + off;
    off = (off + bytes + 255) & ~(size_t)255;
    return p;
  };
  bf16_t* Hb = (bf16_t*)alloc((size_t)N_NODES * HIDDEN * 2);
  bf16_t* Mb = (bf16_t*)alloc((size_t)N_NODES * HIDDEN * 2);
  float*  Ab = (float*)alloc((size_t)N_NODES * HIDDEN * 4);
  bf16_t *Wb[2], *Wihb[2], *Whhb[2];
  for (int l = 0; l < 2; ++l) {
    Wb[l]   = (bf16_t*)alloc((size_t)HIDDEN * HIDDEN * 2);
    Wihb[l] = (bf16_t*)alloc((size_t)3 * HIDDEN * HIDDEN * 2);
    Whhb[l] = (bf16_t*)alloc((size_t)3 * HIDDEN * HIDDEN * 2);
  }

  pad_kernel<<<(N_NODES * HIDDEN) / 256, 256, 0, stream>>>(features, Hb);
  for (int l = 0; l < 2; ++l) {
    cvt_kernel<<<(HIDDEN * HIDDEN) / 256, 256, 0, stream>>>(W[l], Wb[l], HIDDEN * HIDDEN);
    cvt_kernel<<<(3 * HIDDEN * HIDDEN) / 256, 256, 0, stream>>>(wih[l], Wihb[l], 3 * HIDDEN * HIDDEN);
    cvt_kernel<<<(3 * HIDDEN * HIDDEN) / 256, 256, 0, stream>>>(whh[l], Whhb[l], 3 * HIDDEN * HIDDEN);
  }

  for (int l = 0; l < 2; ++l) {
    for (int s = 0; s < N_STEPS; ++s) {
      gemm_m_kernel<<<N_NODES / 80, 256, 0, stream>>>(Hb, Wb[l], b[l], Mb);
      zero_kernel<<<(N_NODES * HIDDEN / 4) / 256, 256, 0, stream>>>((float4*)Ab);
      scatter_kernel<<<(N_EDGES * 16) / 256, 256, 0, stream>>>(Mb, src, dst, Ab);
      gru_kernel<<<N_NODES / 32, 256, 0, stream>>>(Ab, Hb, Wihb[l], Whhb[l], bih[l], bhh[l]);
    }
  }
  head_kernel<<<(N_NODES + 255) / 256, 256, 0, stream>>>(Hb, fc_w, fc_b, (float*)d_out);
}
